// MultiHeadAttention_30167850287106
// MI455X (gfx1250) — matrix operand order repaired and checked by reading, not compiled, not run on hardware
//
#include <hip/hip_runtime.h>

// ---------------------------------------------------------------------------
// MI455X (gfx1250) multi-head attention forward, bf16 WMMA (16x16x32) pipeline
// B=4, S=1024, D=1024, H=16, HD=64
// ---------------------------------------------------------------------------

typedef unsigned short u16;
typedef unsigned int   u32;
typedef __attribute__((ext_vector_type(4)))  u32    u32x4;
typedef __attribute__((ext_vector_type(4)))  float  f32x4;
typedef __attribute__((ext_vector_type(8)))  float  f32x8;
typedef __attribute__((ext_vector_type(8)))  __bf16 bf16x8;
typedef __attribute__((ext_vector_type(16))) __bf16 bf16x16;

union Frag {
    bf16x16 v;
    bf16x8  h[2];
    u16     u[16];
};

__device__ __forceinline__ u16 f2bf(float f) {
    u32 u = __float_as_uint(f);
    u += 0x7fffu + ((u >> 16) & 1u);   // round-to-nearest-even
    return (u16)(u >> 16);
}

__device__ __forceinline__ f32x8 zero8() {
    f32x8 z;
#pragma unroll
    for (int i = 0; i < 8; ++i) z[i] = 0.0f;
    return z;
}

__device__ __forceinline__ f32x8 wmma_bf16(const Frag& a, const Frag& b, f32x8 c) {
    // v_wmma_f32_16x16x32_bf16: D = A(16x32) * B(32x16) + C(16x16 f32)
    return __builtin_amdgcn_wmma_f32_16x16x32_bf16(
        false, a.v, false, b.v, (short)0, c, false, false);
}

// ---------------------------------------------------------------------------
// fp32 -> bf16 bulk conversion
// ---------------------------------------------------------------------------
__global__ __launch_bounds__(256) void k_f32_to_bf16(
    const float* __restrict__ src, u16* __restrict__ dst, int n)
{
    int i = blockIdx.x * 256 + threadIdx.x;
    if (i < n) dst[i] = f2bf(src[i]);
}

// ---------------------------------------------------------------------------
// content_bias[b,h,s] = x[b,s,:] . Wb[:,h] + mask[b,s]   (one WG per (b,s))
// ---------------------------------------------------------------------------
__global__ __launch_bounds__(256) void k_content_bias(
    const float* __restrict__ x, const float* __restrict__ Wb,
    const float* __restrict__ mask, float* __restrict__ cb)
{
    __shared__ float xs[1024];
    __shared__ float partial[16][17];
    const int blk = blockIdx.x;          // 0..4095
    const int b = blk >> 10, s = blk & 1023;
    const float* xrow = x + (size_t)blk * 1024;
    for (int i = threadIdx.x; i < 1024; i += 256) xs[i] = xrow[i];
    __syncthreads();
    const int h  = threadIdx.x >> 4;     // 0..15
    const int ch = threadIdx.x & 15;     // 0..15 (64-wide k chunk)
    float p = 0.0f;
    const int k0 = ch * 64;
#pragma unroll 4
    for (int k = k0; k < k0 + 64; ++k) p += xs[k] * Wb[k * 16 + h];
    partial[h][ch] = p;
    __syncthreads();
    if (threadIdx.x < 16) {
        const int hh = threadIdx.x;
        float sum = 0.0f;
#pragma unroll
        for (int c = 0; c < 16; ++c) sum += partial[hh][c];
        cb[((size_t)b * 16 + hh) * 1024 + s] = sum + mask[b * 1024 + s];
    }
}

// ---------------------------------------------------------------------------
// Tiled bf16 GEMM: C[4096x1024] = A[4096x1024] * Bw[1024x1024]
// WG tile 128x128, K-chunk 64, 8 waves (4m x 2n), each wave 32x64 (2x4 tiles).
// Epilogue selects destination/precision per mode.
// ---------------------------------------------------------------------------
enum { MODE_Q = 0, MODE_K = 1, MODE_V = 2, MODE_DENSE = 3 };

template <int MODE>
__global__ __launch_bounds__(256) void k_gemm_bf16(
    const u16* __restrict__ A,        // [4096][1024] bf16 row-major
    const u16* __restrict__ Bw,       // [1024][1024] bf16 row-major
    const float* __restrict__ mixing, // [16][64]   (MODE_Q)
    const float* __restrict__ bd,     // [1024]     (MODE_DENSE)
    u16* __restrict__ out_bf,         // Q/K/V dest
    float* __restrict__ out_f32)      // dense dest
{
    __shared__ u16 lA[128][72];   // [m][k]  (+8 halves pad: 16B-aligned rows, conflict-free)
    __shared__ u16 lB[128][72];   // [n][k]  (B staged K-major = transposed)

    const int m0 = blockIdx.x * 128, n0 = blockIdx.y * 128;
    const int t = threadIdx.x;
    const int w = t >> 5, lane = t & 31, lm = lane & 15, g = lane >> 4;
    const int wm = w >> 1, wn = w & 1;

    f32x8 acc[2][4];
#pragma unroll
    for (int mt = 0; mt < 2; ++mt)
#pragma unroll
        for (int nt = 0; nt < 4; ++nt) acc[mt][nt] = zero8();

    for (int kc = 0; kc < 1024; kc += 64) {
        __syncthreads();
        {   // stage A tile 128x64 (row-major copy)
            const int r = t >> 1, kh = (t & 1) * 32;
            const u32x4* src = (const u32x4*)(A + (size_t)(m0 + r) * 1024 + kc + kh);
            u32x4* dst = (u32x4*)&lA[r][kh];
            dst[0] = src[0]; dst[1] = src[1]; dst[2] = src[2]; dst[3] = src[3];
            if (kc + 64 < 1024)
                __builtin_prefetch(A + (size_t)(m0 + r) * 1024 + kc + 64 + kh, 0, 1);
        }
        {   // stage B tile 64x128, transposed into [n][k]
            const int k = t >> 2, nh = (t & 3) * 32;
            union { u32x4 q[4]; u16 u[32]; } tmp;
            const u32x4* src = (const u32x4*)(Bw + (size_t)(kc + k) * 1024 + n0 + nh);
            tmp.q[0] = src[0]; tmp.q[1] = src[1]; tmp.q[2] = src[2]; tmp.q[3] = src[3];
#pragma unroll
            for (int j = 0; j < 32; ++j) lB[nh + j][k] = tmp.u[j];
        }
        __syncthreads();

#pragma unroll
        for (int ks = 0; ks < 2; ++ks) {
            Frag af[2];
#pragma unroll
            for (int mt = 0; mt < 2; ++mt) {
                // A element j <-> k = (j&7) + 16*(j>>3) + 8*g (+32*ks)
                const u16* p = &lA[wm * 32 + mt * 16 + lm][0];
                af[mt].h[0] = *(const bf16x8*)(p + 8 * g + 32 * ks);
                af[mt].h[1] = *(const bf16x8*)(p + 16 + 8 * g + 32 * ks);
            }
#pragma unroll
            for (int nt = 0; nt < 4; ++nt) {
                // B element j <-> k = (j&7) + 16*(j>>3) + 8*g (+32*ks) in K-major LDS
                Frag bf;
                const u16* p = &lB[wn * 64 + nt * 16 + lm][0];
                bf.h[0] = *(const bf16x8*)(p + 8 * g + 32 * ks);
                bf.h[1] = *(const bf16x8*)(p + 16 + 8 * g + 32 * ks);
#pragma unroll
                for (int mt = 0; mt < 2; ++mt)
                    acc[mt][nt] = wmma_bf16(af[mt], bf, acc[mt][nt]);
            }
        }
    }

    // epilogue: C/D element i <-> m = i + 8*g, n = lane&15
#pragma unroll
    for (int mt = 0; mt < 2; ++mt)
#pragma unroll
        for (int nt = 0; nt < 4; ++nt) {
            const int col = n0 + wn * 64 + nt * 16 + lm;
#pragma unroll
            for (int i = 0; i < 8; ++i) {
                const int row = m0 + wm * 32 + mt * 16 + i + 8 * g;
                float v = acc[mt][nt][i];
                if (MODE == MODE_Q) {
                    const int hh = col >> 6, hd = col & 63;
                    v *= mixing[hh * 64 + hd] * 0.125f;   // fold 1/sqrt(64)
                    out_bf[(((size_t)(row >> 10) * 16 + hh) * 1024 + (row & 1023)) * 64 + hd] = f2bf(v);
                } else if (MODE == MODE_K) {
                    const int hh = col >> 6, hd = col & 63;
                    out_bf[(((size_t)(row >> 10) * 16 + hh) * 1024 + (row & 1023)) * 64 + hd] = f2bf(v);
                } else if (MODE == MODE_V) {
                    // V stored transposed: [b,h,hd,s]
                    const int hh = col >> 6, hd = col & 63;
                    out_bf[(((size_t)(row >> 10) * 16 + hh) * 64 + hd) * 1024 + (row & 1023)] = f2bf(v);
                } else {
                    out_f32[(size_t)row * 1024 + col] = v + bd[col];
                }
            }
        }
}

// ---------------------------------------------------------------------------
// scores + softmax, fused. One WG = (b, h, 16 query rows); 8 waves cover all
// 1024 keys (wave w -> keys [w*128, w*128+128)). K-depth = 64 (2 WMMA steps).
// Writes attn_weights fp32 straight into d_out.
// ---------------------------------------------------------------------------
__global__ __launch_bounds__(256) void k_scores_softmax(
    const u16* __restrict__ Qm,   // [B,H,S,64] bf16 (mixing & 1/8 folded in)
    const u16* __restrict__ Kb,   // [B,H,S,64] bf16
    const float* __restrict__ cb, // [B,H,S] fp32 (mask folded in)
    float* __restrict__ P)        // [B,H,S,S] fp32
{
    __shared__ float red[16][8];
    const int q0 = blockIdx.x * 16, h = blockIdx.y, b = blockIdx.z;
    const int t = threadIdx.x, w = t >> 5, lane = t & 31, lm = lane & 15, g = lane >> 4;
    const size_t head = ((size_t)(b * 16 + h)) * 1024 * 64;
    const int nbase = w * 128;

    // A fragments: rows q0..q0+15, direct global loads (two 16B runs per frag)
    Frag a[2];
    const u16* arow = Qm + head + (size_t)(q0 + lm) * 64;
#pragma unroll
    for (int ks = 0; ks < 2; ++ks) {
        a[ks].h[0] = *(const bf16x8*)(arow + 8 * g + 32 * ks);
        a[ks].h[1] = *(const bf16x8*)(arow + 16 + 8 * g + 32 * ks);
    }

    f32x8 acc[8];
#pragma unroll
    for (int nt = 0; nt < 8; ++nt) {
        acc[nt] = zero8();
        const u16* brow = Kb + head + (size_t)(nbase + nt * 16 + lm) * 64;
#pragma unroll
        for (int ks = 0; ks < 2; ++ks) {
            Frag bf;
            bf.h[0] = *(const bf16x8*)(brow + 8 * g + 32 * ks);
            bf.h[1] = *(const bf16x8*)(brow + 16 + 8 * g + 32 * ks);
            acc[nt] = wmma_bf16(a[ks], bf, acc[nt]);
        }
        const float c = cb[((size_t)(b * 16 + h)) * 1024 + nbase + nt * 16 + lm];
#pragma unroll
        for (int i = 0; i < 8; ++i) acc[nt][i] += c;
    }

    // ---- row max (row m = i + 8*g lives in 16-lane half g) ----
    float fullmax[8];
    {
        float rmax[8];
#pragma unroll
        for (int i = 0; i < 8; ++i) {
            float m = acc[0][i];
#pragma unroll
            for (int nt = 1; nt < 8; ++nt) m = fmaxf(m, acc[nt][i]);
            for (int s = 1; s < 16; s <<= 1) m = fmaxf(m, __shfl_xor(m, s, 32));
            rmax[i] = m;
        }
        if (lm == 0) {
#pragma unroll
            for (int i = 0; i < 8; ++i) red[i + 8 * g][w] = rmax[i];
        }
        __syncthreads();
#pragma unroll
        for (int i = 0; i < 8; ++i) {
            float m = -3.402823466e38f;
#pragma unroll
            for (int wv = 0; wv < 8; ++wv) m = fmaxf(m, red[i + 8 * g][wv]);
            fullmax[i] = m;
        }
        __syncthreads();
    }

    // ---- exp & row sum ----
    float inv[8];
    {
        float rsum[8];
#pragma unroll
        for (int i = 0; i < 8; ++i) rsum[i] = 0.0f;
#pragma unroll
        for (int nt = 0; nt < 8; ++nt)
#pragma unroll
            for (int i = 0; i < 8; ++i) {
                const float e = __expf(acc[nt][i] - fullmax[i]);
                acc[nt][i] = e;
                rsum[i] += e;
            }
#pragma unroll
        for (int i = 0; i < 8; ++i)
            for (int s = 1; s < 16; s <<= 1) rsum[i] += __shfl_xor(rsum[i], s, 32);
        if (lm == 0) {
#pragma unroll
            for (int i = 0; i < 8; ++i) red[i + 8 * g][w] = rsum[i];
        }
        __syncthreads();
#pragma unroll
        for (int i = 0; i < 8; ++i) {
            float sum = 0.0f;
#pragma unroll
            for (int wv = 0; wv < 8; ++wv) sum += red[i + 8 * g][wv];
            inv[i] = 1.0f / sum;
        }
    }

    // ---- store normalized probabilities ----
#pragma unroll
    for (int nt = 0; nt < 8; ++nt)
#pragma unroll
        for (int i = 0; i < 8; ++i)
            P[((size_t)(b * 16 + h) * 1024 + q0 + i + 8 * g) * 1024 + nbase + nt * 16 + lm]
                = acc[nt][i] * inv[i];
}

// ---------------------------------------------------------------------------
// attn_output_head = P @ V.  A = P fp32 (converted to bf16 in regs),
// B = transposed V [b,h,64,1024] -> contiguous fragments. WG: 128 q-rows x 64.
// Output bf16 in [b,s,h*64+hd] layout for the final dense GEMM.
// ---------------------------------------------------------------------------
__global__ __launch_bounds__(256) void k_attn_v(
    const float* __restrict__ P,   // [B,H,S,S]
    const u16* __restrict__ Vt,    // [B,H,64,S]
    u16* __restrict__ O)           // [B,S,1024]
{
    const int q0 = blockIdx.x * 128, h = blockIdx.y, b = blockIdx.z;
    const int t = threadIdx.x, w = t >> 5, lane = t & 31, lm = lane & 15, g = lane >> 4;
    const int qrow = q0 + w * 16 + lm;
    const float* prow = P + ((size_t)(b * 16 + h) * 1024 + qrow) * 1024;
    const size_t vbase = ((size_t)(b * 16 + h)) * 64 * 1024;

    f32x8 acc[4];
#pragma unroll
    for (int nt = 0; nt < 4; ++nt) acc[nt] = zero8();

    for (int ks = 0; ks < 32; ++ks) {
        Frag af;
        const float* p0 = prow + 8 * g + 32 * ks;
        f32x4 r0 = *(const f32x4*)(p0);
        f32x4 r1 = *(const f32x4*)(p0 + 4);
        f32x4 r2 = *(const f32x4*)(p0 + 16);
        f32x4 r3 = *(const f32x4*)(p0 + 20);
#pragma unroll
        for (int j = 0; j < 4; ++j) {
            af.u[j]      = f2bf(r0[j]);
            af.u[4 + j]  = f2bf(r1[j]);
            af.u[8 + j]  = f2bf(r2[j]);
            af.u[12 + j] = f2bf(r3[j]);
        }
#pragma unroll
        for (int nt = 0; nt < 4; ++nt) {
            const u16* vrow = Vt + vbase + (size_t)(nt * 16 + lm) * 1024;
            Frag bf;
            bf.h[0] = *(const bf16x8*)(vrow + 8 * g + 32 * ks);
            bf.h[1] = *(const bf16x8*)(vrow + 16 + 8 * g + 32 * ks);
            acc[nt] = wmma_bf16(af, bf, acc[nt]);
        }
    }

#pragma unroll
    for (int nt = 0; nt < 4; ++nt)
#pragma unroll
        for (int i = 0; i < 8; ++i)
            O[((size_t)b * 1024 + q0 + w * 16 + i + 8 * g) * 1024 + h * 64 + nt * 16 + lm]
                = f2bf(acc[nt][i]);
}

// ---------------------------------------------------------------------------
// Host launcher
// ---------------------------------------------------------------------------
extern "C" void kernel_launch(void* const* d_in, const int* in_sizes, int n_in,
                              void* d_out, int out_size, void* d_ws, size_t ws_size,
                              hipStream_t stream)
{
    (void)in_sizes; (void)n_in; (void)out_size; (void)ws_size;

    const float* x      = (const float*)d_in[0];   // [4,1024,1024]
    const float* mask   = (const float*)d_in[1];   // [4,1,1,1024]
    const float* Wq     = (const float*)d_in[2];
    const float* Wk     = (const float*)d_in[3];
    const float* Wv     = (const float*)d_in[4];
    const float* Wb     = (const float*)d_in[5];   // [1024,16]
    const float* mixing = (const float*)d_in[6];   // [16,64]
    const float* Wd     = (const float*)d_in[7];
    const float* bd     = (const float*)d_in[8];   // [1024]

    float* out      = (float*)d_out;
    float* attn_out = out;                              // [4,1024,1024]
    float* attn_w   = out + (size_t)4 * 1024 * 1024;    // [4,16,1024,1024]

    // workspace carve-up (~48.3 MB total)
    char* ws = (char*)d_ws;
    size_t o = 0;
    u16*   xbf  = (u16*)(ws + o); o += (size_t)4096 * 1024 * 2;
    u16*   wqbf = (u16*)(ws + o); o += (size_t)1024 * 1024 * 2;
    u16*   wkbf = (u16*)(ws + o); o += (size_t)1024 * 1024 * 2;
    u16*   wvbf = (u16*)(ws + o); o += (size_t)1024 * 1024 * 2;
    u16*   wdbf = (u16*)(ws + o); o += (size_t)1024 * 1024 * 2;
    u16*   Qm   = (u16*)(ws + o); o += (size_t)4096 * 1024 * 2;  // [B,H,S,64]
    u16*   Kb   = (u16*)(ws + o); o += (size_t)4096 * 1024 * 2;  // [B,H,S,64]
    u16*   Vt   = (u16*)(ws + o); o += (size_t)4096 * 1024 * 2;  // [B,H,64,S]
    float* cbv  = (float*)(ws + o); o += (size_t)4 * 16 * 1024 * 4;
    u16*   abf  = (u16*)(ws + o); o += (size_t)4096 * 1024 * 2;  // [B,S,1024]

    // 1) precision conversion
    k_f32_to_bf16<<<dim3(4194304 / 256), 256, 0, stream>>>(x,  xbf,  4194304);
    k_f32_to_bf16<<<dim3(1048576 / 256), 256, 0, stream>>>(Wq, wqbf, 1048576);
    k_f32_to_bf16<<<dim3(1048576 / 256), 256, 0, stream>>>(Wk, wkbf, 1048576);
    k_f32_to_bf16<<<dim3(1048576 / 256), 256, 0, stream>>>(Wv, wvbf, 1048576);
    k_f32_to_bf16<<<dim3(1048576 / 256), 256, 0, stream>>>(Wd, wdbf, 1048576);

    // 2) content bias (+mask)
    k_content_bias<<<dim3(4096), 256, 0, stream>>>(x, Wb, mask, cbv);

    // 3) QKV projections (mixing & 1/sqrt(hd) folded into Q; V transposed)
    k_gemm_bf16<MODE_Q><<<dim3(32, 8), 256, 0, stream>>>(xbf, wqbf, mixing, nullptr, Qm, nullptr);
    k_gemm_bf16<MODE_K><<<dim3(32, 8), 256, 0, stream>>>(xbf, wkbf, nullptr, nullptr, Kb, nullptr);
    k_gemm_bf16<MODE_V><<<dim3(32, 8), 256, 0, stream>>>(xbf, wvbf, nullptr, nullptr, Vt, nullptr);

    // 4) fused scores + softmax -> attn_weights (d_out region 2)
    k_scores_softmax<<<dim3(64, 16, 4), 256, 0, stream>>>(Qm, Kb, cbv, attn_w);

    // 5) P @ V -> bf16 context in [B,S,D] layout
    k_attn_v<<<dim3(8, 16, 4), 256, 0, stream>>>(attn_w, Vt, abf);

    // 6) final dense projection + bias -> attn_output (d_out region 1)
    k_gemm_bf16<MODE_DENSE><<<dim3(32, 8), 256, 0, stream>>>(abf, wdbf, nullptr, bd, nullptr, attn_out);
}
